// CustomLSTMModel_75179107549339
// MI455X (gfx1250) — compile-verified, matrix operand order checked by reading, not run on hardware
//
#include <hip/hip_runtime.h>
#include <hip/hip_bf16.h>

// ---------------------------------------------------------------------------
// Types for CDNA5 WMMA (wave32): 16x16x32 bf16 -> f32 accumulate
// ---------------------------------------------------------------------------
typedef __attribute__((ext_vector_type(16))) __bf16 v16bf;
typedef __attribute__((ext_vector_type(8)))  float  v8f;

__device__ __forceinline__ v8f wmma_bf16(v16bf a, v16bf b, v8f c) {
    // (neg_a, A, neg_b, B, c_mod, C, reuse_a, reuse_b)
    return __builtin_amdgcn_wmma_f32_16x16x32_bf16(
        false, a, false, b, (short)0, c, false, false);
}

__device__ __forceinline__ float sigmoid_(float x) {
    return 1.0f / (1.0f + __expf(-x));
}
__device__ __forceinline__ float tanh_(float x) {
    float e = __expf(-2.0f * x);
    return (1.0f - e) / (1.0f + e);
}

// Async copy of 16 bytes global -> LDS (gfx1250 GLOBAL_LOAD_ASYNC_TO_LDS_B128,
// GV addressing; tracked by ASYNCcnt). Inline asm: portable across both
// toolchains (the clang builtin arity differs between them).
__device__ __forceinline__ void async_copy_b128(void* lds_dst, const void* gsrc) {
    unsigned ldsoff = (unsigned)(uintptr_t)lds_dst;   // low 32 bits = LDS offset
    asm volatile("global_load_async_to_lds_b128 %0, %1, off"
                 :: "v"(ldsoff), "v"(gsrc) : "memory");
}
__device__ __forceinline__ void wait_asynccnt0() {
    asm volatile("s_wait_asynccnt 0x0" ::: "memory");
}

static constexpr int B_  = 32;
static constexpr int S_  = 2048;
static constexpr int D_  = 256;
static constexpr int Hn_ = 256;
static constexpr int G4_ = 4 * Hn_;            // 1024 gate columns
static constexpr int M_  = B_ * S_;            // 65536 GEMM rows

// ---------------------------------------------------------------------------
// fp32 -> bf16 elementwise conversion
// ---------------------------------------------------------------------------
__global__ void f32_to_bf16_kernel(const float* __restrict__ src,
                                   __bf16* __restrict__ dst, int n) {
    int i = blockIdx.x * blockDim.x + threadIdx.x;
    if (i < n) dst[i] = (__bf16)src[i];
}

// Build transposed, gate-concatenated bf16 weights:
// dst[n*256 + k] = W_gate[k*256 + j]  with n = gate*256 + j   (n in [0,1024))
__global__ void pack_wT_kernel(const float* __restrict__ g0,
                               const float* __restrict__ g1,
                               const float* __restrict__ g2,
                               const float* __restrict__ g3,
                               __bf16* __restrict__ dst) {
    int idx = blockIdx.x * blockDim.x + threadIdx.x;   // 1024*256 total
    if (idx >= G4_ * 256) return;
    int n = idx >> 8;          // column of original [256,1024] matrix
    int k = idx & 255;         // row (K index)
    int gate = n >> 8;
    int j    = n & 255;
    const float* s = (gate == 0) ? g0 : (gate == 1) ? g1 : (gate == 2) ? g2 : g3;
    dst[idx] = (__bf16)s[k * 256 + j];
}

// ---------------------------------------------------------------------------
// Phase 1: x_proj[s][b][n] = inputs[b,s,:] @ Wx[:,n] + bias[n]
// Xb  : bf16 [65536, 256]   (row m = b*2048 + s)
// WxT : bf16 [1024, 256]    (row n holds column n of Wx, contiguous in K)
// out : fp32 [2048][32][1024]
// Block: 256 threads (8 waves). The block's 64-column B panel (32 KB, full K
// range) is staged once into LDS with async-to-LDS copies; every wave then
// feeds WMMA B operands from LDS while streaming its exclusive A rows from
// global, removing the 8x-redundant B traffic.
// ---------------------------------------------------------------------------
__global__ __launch_bounds__(256) void xproj_kernel(
    const __bf16* __restrict__ Xb,
    const __bf16* __restrict__ WxT,
    const float* __restrict__ bi, const float* __restrict__ bf_,
    const float* __restrict__ bo, const float* __restrict__ bc,
    float* __restrict__ xproj) {

    __shared__ __attribute__((aligned(32))) __bf16 Blds[64 * 256];   // 32 KB

    const int tid   = threadIdx.x;
    const int lane  = tid & 31;
    const int wave  = tid >> 5;
    const int mBase = blockIdx.x * 128 + wave * 16;
    const int nBase = blockIdx.y * 64;
    const int lr = lane & 15;    // row-in-tile (A) / col-in-tile (B)
    const int hi = lane >> 4;    // selects K half (contiguous k = 16*hi + s, same for A and B)

    // ---- Stage B panel: rows [nBase, nBase+64) of WxT, 32768 bytes ----
    {
        const char* src = (const char*)(WxT + (size_t)nBase * 256);
        char*       dst = (char*)Blds;
        #pragma unroll
        for (int it = 0; it < 8; ++it) {
            int off = (it * 256 + tid) * 16;
            async_copy_b128(dst + off, src + off);
        }
        wait_asynccnt0();
        __syncthreads();
    }

    const v8f vz = {0.f,0.f,0.f,0.f,0.f,0.f,0.f,0.f};
    v8f acc0 = vz, acc1 = vz, acc2 = vz, acc3 = vz;

    #pragma unroll
    for (int k = 0; k < 256; k += 32) {
        v16bf a  = *(const v16bf*)(Xb + (size_t)(mBase + lr) * 256 + k + hi * 16);
        v16bf b0 = *(const v16bf*)(&Blds[( 0 + lr) * 256 + k + hi * 16]);
        v16bf b1 = *(const v16bf*)(&Blds[(16 + lr) * 256 + k + hi * 16]);
        v16bf b2 = *(const v16bf*)(&Blds[(32 + lr) * 256 + k + hi * 16]);
        v16bf b3 = *(const v16bf*)(&Blds[(48 + lr) * 256 + k + hi * 16]);
        acc0 = wmma_bf16(a, b0, acc0);
        acc1 = wmma_bf16(a, b1, acc1);
        acc2 = wmma_bf16(a, b2, acc2);
        acc3 = wmma_bf16(a, b3, acc3);
    }

    // Epilogue: C/D layout -> element (r, lane): row = r + 8*hi, col = lr
    #pragma unroll
    for (int nt = 0; nt < 4; ++nt) {
        const v8f acc = (nt == 0) ? acc0 : (nt == 1) ? acc1 : (nt == 2) ? acc2 : acc3;
        int col  = nBase + nt * 16 + lr;
        int gate = col >> 8, j = col & 255;
        float bias = ((gate == 0) ? bi : (gate == 1) ? bf_ : (gate == 2) ? bo : bc)[j];
        #pragma unroll
        for (int r = 0; r < 8; ++r) {
            int m = mBase + r + 8 * hi;
            int b = m >> 11;          // m / 2048
            int s = m & 2047;         // m % 2048
            xproj[(size_t)s * (B_ * G4_) + b * G4_ + col] = acc[r] + bias;
        }
    }
}

// ---------------------------------------------------------------------------
// Phase 2: persistent single-WGP LSTM scan.
// 1024 threads = 32 waves. Wave (mTile in {0,1}, hTile in {0..15}) owns the
// 16x16 (batch, hidden) tile at (mTile*16, hTile*16) and computes all four
// gates for it, so the cell update is pure register math. C stays in VGPRs
// for all 2048 steps; H(t) is shared via 16 KB of bf16 LDS. The i/f halves of
// WhT (256 KB) are staged into LDS once with async copies (WGP has 320 KB);
// o/c halves stream from L2. pred = H . fc_w is fused via shuffles + LDS
// atomics. Dynamic LDS: 256K (WhLds) + 16K (Hlds) + 128B (pred_acc).
// ---------------------------------------------------------------------------
__global__ __launch_bounds__(1024) void lstm_scan_kernel(
    const float* __restrict__ xproj,   // [2048][32][1024] (bias included)
    const __bf16* __restrict__ WhT,    // [1024][256]
    const float* __restrict__ H0,      // [32,256]
    const float* __restrict__ C0,      // [32,256]
    const float* __restrict__ fc_w,    // [256]
    const float* __restrict__ fc_b,    // [1]
    float* __restrict__ out) {         // pred[65536] | Hf[8192] | Cf[8192]

    extern __shared__ __attribute__((aligned(32))) char smem[];
    __bf16* WhLds    = (__bf16*)smem;                      // [512*256] 256 KB
    __bf16* Hlds     = (__bf16*)(smem + 512 * 256 * 2);    // [32*256]   16 KB
    float*  pred_acc = (float*)(smem + 512 * 256 * 2 + B_ * Hn_ * 2);

    const int tid   = threadIdx.x;
    const int lane  = tid & 31;
    const int wave  = tid >> 5;
    const int mTile = wave & 1;
    const int hTile = wave >> 1;
    const int lr = lane & 15;
    const int hi = lane >> 4;
    const int h  = hTile * 16 + lr;        // hidden column owned by this lane
    const float fw  = fc_w[h];
    const float fcb = fc_b[0];

    // ---- One-time async stage: WhT rows [0,512) -> LDS (gates i, f) ----
    {
        const char* src = (const char*)WhT;
        char*       dst = (char*)WhLds;
        #pragma unroll
        for (int it = 0; it < 16; ++it) {
            int off = (it * 1024 + tid) * 16;   // 262144 bytes total
            async_copy_b128(dst + off, src + off);
        }
        wait_asynccnt0();
    }

    float C[8], H[8];
    #pragma unroll
    for (int r = 0; r < 8; ++r) {
        int b = mTile * 16 + r + 8 * hi;
        C[r] = C0[b * Hn_ + h];
        H[r] = 0.f;
    }
    for (int i = tid; i < B_ * Hn_; i += 1024) Hlds[i] = (__bf16)H0[i];
    if (tid < B_) pred_acc[tid] = 0.f;
    __syncthreads();

    const v8f vz = {0.f,0.f,0.f,0.f,0.f,0.f,0.f,0.f};

    for (int t = 0; t < S_; ++t) {
        const float* xp = xproj + (size_t)t * (B_ * G4_);

        // Seed accumulators with x-projection (bias already folded in).
        v8f acc0 = vz, acc1 = vz, acc2 = vz, acc3 = vz;
        #pragma unroll
        for (int r = 0; r < 8; ++r) {
            int b = mTile * 16 + r + 8 * hi;
            const float* row = xp + b * G4_;
            acc0[r] = row[0 * Hn_ + h];
            acc1[r] = row[1 * Hn_ + h];
            acc2[r] = row[2 * Hn_ + h];
            acc3[r] = row[3 * Hn_ + h];
        }

        // g += H(t-1) @ Wh : A from LDS H; B for i/f from LDS, o/c from L2.
        #pragma unroll
        for (int k = 0; k < Hn_; k += 32) {
            v16bf a  = *(const v16bf*)(&Hlds[(mTile * 16 + lr) * Hn_ + k + hi * 16]);
            v16bf b0 = *(const v16bf*)(&WhLds[(0 * Hn_ + h) * Hn_ + k + hi * 16]);
            v16bf b1 = *(const v16bf*)(&WhLds[(1 * Hn_ + h) * Hn_ + k + hi * 16]);
            v16bf b2 = *(const v16bf*)(WhT + (size_t)(2 * Hn_ + h) * Hn_ + k + hi * 16);
            v16bf b3 = *(const v16bf*)(WhT + (size_t)(3 * Hn_ + h) * Hn_ + k + hi * 16);
            acc0 = wmma_bf16(a, b0, acc0);
            acc1 = wmma_bf16(a, b1, acc1);
            acc2 = wmma_bf16(a, b2, acc2);
            acc3 = wmma_bf16(a, b3, acc3);
        }
        __syncthreads();   // all H(t-1) reads done before overwrite

        // Cell update in registers; publish H(t) to LDS; pred partials.
        float partial[8];
        #pragma unroll
        for (int r = 0; r < 8; ++r) {
            float gi = sigmoid_(acc0[r]);
            float gf = sigmoid_(acc1[r]);
            float go = sigmoid_(acc2[r]);
            float gc = tanh_(acc3[r]);
            float c  = gf * C[r] + gi * gc;
            C[r] = c;
            float hv = go * tanh_(c);
            H[r] = hv;
            int b = mTile * 16 + r + 8 * hi;
            Hlds[b * Hn_ + h] = (__bf16)hv;
            partial[r] = hv * fw;
        }

        // Reduce over the 16 lanes of each half (same batch set, different h).
        #pragma unroll
        for (int r = 0; r < 8; ++r) {
            float p = partial[r];
            p += __shfl_xor(p, 1);
            p += __shfl_xor(p, 2);
            p += __shfl_xor(p, 4);
            p += __shfl_xor(p, 8);
            partial[r] = p;
        }
        if (lr == 0) {
            #pragma unroll
            for (int r = 0; r < 8; ++r) {
                int b = mTile * 16 + r + 8 * hi;
                atomicAdd(&pred_acc[b], partial[r]);
            }
        }

        // Prefetch next step's x_proj slice (global_prefetch_b8).
        if (t + 1 < S_) __builtin_prefetch(xp + (B_ * G4_) + tid * 32, 0, 1);

        __syncthreads();   // H(t) + pred_acc complete
        if (tid < B_) {
            out[tid * S_ + t] = pred_acc[tid] + fcb;
            pred_acc[tid] = 0.f;
        }
        __syncthreads();   // zeroing visible before next step's atomics
    }

    // Final states.
    #pragma unroll
    for (int r = 0; r < 8; ++r) {
        int b = mTile * 16 + r + 8 * hi;
        out[B_ * S_            + b * Hn_ + h] = H[r];
        out[B_ * S_ + B_ * Hn_ + b * Hn_ + h] = C[r];
    }
}

// ---------------------------------------------------------------------------
// Launch
// ---------------------------------------------------------------------------
extern "C" void kernel_launch(void* const* d_in, const int* in_sizes, int n_in,
                              void* d_out, int out_size, void* d_ws, size_t ws_size,
                              hipStream_t stream) {
    const float* inputs = (const float*)d_in[0];
    const float* H0     = (const float*)d_in[1];
    const float* C0     = (const float*)d_in[2];
    const float* Wxi = (const float*)d_in[3];
    const float* Whi = (const float*)d_in[4];
    const float* bi  = (const float*)d_in[5];
    const float* Wxf = (const float*)d_in[6];
    const float* Whf = (const float*)d_in[7];
    const float* bf_ = (const float*)d_in[8];
    const float* Wxo = (const float*)d_in[9];
    const float* Who = (const float*)d_in[10];
    const float* bo  = (const float*)d_in[11];
    const float* Wxc = (const float*)d_in[12];
    const float* Whc = (const float*)d_in[13];
    const float* bc  = (const float*)d_in[14];
    const float* fcw = (const float*)d_in[15];
    const float* fcb = (const float*)d_in[16];
    float* out = (float*)d_out;

    char* ws = (char*)d_ws;
    const size_t xproj_bytes = (size_t)S_ * B_ * G4_ * sizeof(float);   // 256 MB
    const size_t xb_bytes    = (size_t)M_ * D_ * sizeof(__bf16);        //  32 MB
    const size_t wT_bytes    = (size_t)G4_ * 256 * sizeof(__bf16);      // 512 KB

    float*  xproj = (float*)ws;
    __bf16* Xb    = (__bf16*)(ws + xproj_bytes);
    __bf16* WxT   = (__bf16*)(ws + xproj_bytes + xb_bytes);
    __bf16* WhT   = (__bf16*)(ws + xproj_bytes + xb_bytes + wT_bytes);

    // Conversions / weight packing (stream-ordered).
    {
        int n = M_ * D_;
        f32_to_bf16_kernel<<<(n + 255) / 256, 256, 0, stream>>>(inputs, Xb, n);
    }
    pack_wT_kernel<<<(G4_ * 256 + 255) / 256, 256, 0, stream>>>(Wxi, Wxf, Wxo, Wxc, WxT);
    pack_wT_kernel<<<(G4_ * 256 + 255) / 256, 256, 0, stream>>>(Whi, Whf, Who, Whc, WhT);

    // Phase 1: big WMMA GEMM with async-staged B panel.
    dim3 gA(M_ / 128, G4_ / 64);    // (512, 16)
    xproj_kernel<<<gA, 256, 0, stream>>>(Xb, WxT, bi, bf_, bo, bc, xproj);

    // Phase 2: persistent scan (one WGP), 278656 bytes dynamic LDS.
    const size_t scan_lds = 512 * 256 * 2 + (size_t)B_ * Hn_ * 2 + B_ * sizeof(float);
    lstm_scan_kernel<<<1, 1024, scan_lds, stream>>>(xproj, WhT, H0, C0, fcw, fcb, out);
}
